// SelfAttention_35485019800351
// MI455X (gfx1250) — matrix operand order repaired and checked by reading, not compiled, not run on hardware
//
#include <hip/hip_runtime.h>

typedef _Float16 half_t;
typedef __attribute__((ext_vector_type(16))) _Float16 v16h;
typedef __attribute__((ext_vector_type(8)))  _Float16 v8h;
typedef __attribute__((ext_vector_type(8)))  float    v8f;
typedef __attribute__((ext_vector_type(4)))  unsigned int u32x4;
typedef __attribute__((ext_vector_type(8)))  unsigned int u32x8;

#define EMBED 1024
#define HEADS 16
#define HDIM  64
#define ROWS  16384   // B*S = 4*4096
#define KSTEP 64      // K-slice per double-buffer stage
#define LROW  72      // LDS halves per row: 64 data + 8 pad (TDM pads 16B per 128B row)
#define NIT   (EMBED / KSTEP)

__device__ __forceinline__ v16h cat16(v8h lo, v8h hi) {
  return __builtin_shufflevector(lo, hi, 0,1,2,3,4,5,6,7,8,9,10,11,12,13,14,15);
}

// ---- Tensor Data Mover: async 2D tile load (global -> LDS), wave-level ----
// D# group0: count=1, lds byte addr, 57-bit global addr, type=2 ("image")
// D# group1: data_size=2B, pad_enable, pad_interval=32 DWORDs (code 4),
//            pad_amount=4 DWORDs (code 3), dims/tile/stride in element units.
__device__ __forceinline__ void tdm_load_2d(unsigned int lds_byte_addr,
                                            const void* gaddr,
                                            unsigned int tensor_d0,
                                            unsigned int tensor_d1,
                                            unsigned int tile_d0,
                                            unsigned int tile_d1,
                                            unsigned int d0_stride) {
  unsigned long long ga = (unsigned long long)gaddr;
  u32x4 g0;
  g0[0] = 1u;                                              // count=1, user mode
  g0[1] = lds_byte_addr;                                   // lds_addr [63:32]
  g0[2] = (unsigned int)ga;                                // global_addr low
  g0[3] = (unsigned int)((ga >> 32) & 0x1FFFFFFu)          // global_addr[56:32]
          | 0x80000000u;                                   // type=2 at [127:126]
  u32x8 g1;
  g1[0] = (1u << 16)      // data_size = 2 bytes
        | (1u << 20)      // pad_enable
        | (4u << 22)      // pad_interval: 32 DWORDs (128B rows)
        | (3u << 25);     // pad_amount: 4 DWORDs (16B)
  g1[1] = (tensor_d0 & 0xFFFFu) << 16;                     // tensor_dim0[15:0]
  g1[2] = (tensor_d0 >> 16) | ((tensor_d1 & 0xFFFFu) << 16);
  g1[3] = (tensor_d1 >> 16) | ((tile_d0 & 0xFFFFu) << 16);
  g1[4] = (tile_d1 & 0xFFFFu);                             // tile_dim1, tile_dim2=0
  g1[5] = d0_stride;                                       // tensor_dim0_stride lo
  g1[6] = 0u;
  g1[7] = 0u;
  asm volatile("tensor_load_to_lds %0, %1" :: "s"(g0), "s"(g1) : "memory");
}

// ---------------- fp32 -> f16 conversion ----------------
__global__ __launch_bounds__(256) void cvt_f32_f16(const float* __restrict__ in,
                                                   half_t* __restrict__ out, int n) {
  int i = (blockIdx.x * 256 + threadIdx.x) * 4;
  if (i + 3 < n) {
    float4 v = *(const float4*)(in + i);
    out[i + 0] = (half_t)v.x; out[i + 1] = (half_t)v.y;
    out[i + 2] = (half_t)v.z; out[i + 3] = (half_t)v.w;
  } else {
    for (; i < n; ++i) out[i] = (half_t)in[i];
  }
}

// ---------------- TDM double-buffered WMMA GEMM: C = A * W^T + bias ----------
// A: [ROWS x 1024] f16 row-major, W: [1024 x 1024] f16 row-major (out x in)
// block = 256 threads (8 waves), tile 128(M) x 128(N), K-step 64, 2 LDS buffers
template <bool OUT_F32>
__global__ __launch_bounds__(256) void gemm16k(const half_t* __restrict__ A,
                                               const half_t* __restrict__ W,
                                               const float* __restrict__ bias,
                                               float* __restrict__ Cf,
                                               half_t* __restrict__ Ch) {
  __shared__ __align__(16) half_t As[2][128 * LROW];
  __shared__ __align__(16) half_t Bs[2][128 * LROW];

  const int tid  = threadIdx.x;
  const int lane = tid & 31;
  const int wave = tid >> 5;
  const int hs   = lane >> 4;    // half-wave select
  const int l16  = lane & 15;
  const int m0   = blockIdx.y * 128;
  const int n0   = blockIdx.x * 128;
  const int wm   = (wave & 3) * 32;   // wave M offset inside tile
  const int wn   = (wave >> 2) * 64;  // wave N offset inside tile

  const unsigned int ldsA0 = (unsigned int)(unsigned long long)&As[0][0];
  const unsigned int ldsA1 = (unsigned int)(unsigned long long)&As[1][0];
  const unsigned int ldsB0 = (unsigned int)(unsigned long long)&Bs[0][0];
  const unsigned int ldsB1 = (unsigned int)(unsigned long long)&Bs[1][0];

  v8f acc[2][4] = {};

  // prime buffer 0 with the k0=0 slices of A (128 rows) and W (128 rows)
  if (wave == 0) {
    tdm_load_2d(ldsA0, A + (size_t)m0 * EMBED, EMBED, ROWS, KSTEP, 128, EMBED);
    tdm_load_2d(ldsB0, W + (size_t)n0 * EMBED, EMBED, EMBED, KSTEP, 128, EMBED);
  }
  __builtin_amdgcn_s_wait_tensorcnt(0);
  __syncthreads();

  for (int it = 0; it < NIT; ++it) {
    const int cur = it & 1;
    const half_t* Acur = &As[cur][0];
    const half_t* Bcur = &Bs[cur][0];

    // kick off next K-slice DMA into the other buffer (overlaps with WMMAs)
    if (wave == 0 && it + 1 < NIT) {
      const int kn = (it + 1) * KSTEP;
      tdm_load_2d(cur ? ldsA0 : ldsA1, A + (size_t)m0 * EMBED + kn,
                  EMBED, ROWS, KSTEP, 128, EMBED);
      tdm_load_2d(cur ? ldsB0 : ldsB1, W + (size_t)n0 * EMBED + kn,
                  EMBED, EMBED, KSTEP, 128, EMBED);
    }

#pragma unroll
    for (int ks = 0; ks < 2; ++ks) {
      v16h af[2], bf[4];
#pragma unroll
      for (int mi = 0; mi < 2; ++mi) {
        // A 16x32 f16 as loaded here (contiguous k, same order as B): lane row M=l16; 16 contiguous K at base hs*16
        const half_t* p = Acur + (wm + mi * 16 + l16) * LROW + ks * 32 + hs * 16;
        af[mi] = cat16(*(const v8h*)p, *(const v8h*)(p + 8));
      }
#pragma unroll
      for (int ni = 0; ni < 4; ++ni) {
        // B 32x16 f16 as loaded here (contiguous k, same order as A): lane col N=l16; 16 contiguous K at base hs*16
        const half_t* p = Bcur + (wn + ni * 16 + l16) * LROW + ks * 32 + hs * 16;
        bf[ni] = cat16(*(const v8h*)p, *(const v8h*)(p + 8));
      }
#pragma unroll
      for (int mi = 0; mi < 2; ++mi)
#pragma unroll
        for (int ni = 0; ni < 4; ++ni)
          acc[mi][ni] = __builtin_amdgcn_wmma_f32_16x16x32_f16(
              false, af[mi], false, bf[ni], (short)0, acc[mi][ni], false, false);
    }

    // wave 0 waits for its in-flight DMA; barrier publishes LDS to all waves
    __builtin_amdgcn_s_wait_tensorcnt(0);
    __syncthreads();
  }

  // epilogue: C/D layout -> element r: M = r + hs*8, N = l16
#pragma unroll
  for (int mi = 0; mi < 2; ++mi) {
#pragma unroll
    for (int ni = 0; ni < 4; ++ni) {
      const int gcol = n0 + wn + ni * 16 + l16;
      const float bv = bias[gcol];
#pragma unroll
      for (int r = 0; r < 8; ++r) {
        const int grow = m0 + wm + mi * 16 + hs * 8 + r;
        const float v = acc[mi][ni][r] + bv;
        if (OUT_F32) Cf[(size_t)grow * EMBED + gcol] = v;
        else         Ch[(size_t)grow * EMBED + gcol] = (half_t)v;
      }
    }
  }
}

// ---------------- per-position attention across heads ----------------
// scores[i][j] = (q[p,i,:] . k[p,j,:]) / 8 ; softmax over j ; out = w @ v
__global__ __launch_bounds__(256) void attn_heads(const half_t* __restrict__ Q,
                                                  const half_t* __restrict__ K,
                                                  const half_t* __restrict__ V,
                                                  half_t* __restrict__ O) {
  __shared__ float wt[8][16][17];  // [wave][i][j], padded

  const int tid  = threadIdx.x;
  const int lane = tid & 31;
  const int wave = tid >> 5;
  const int hs   = lane >> 4;
  const int l16  = lane & 15;
  const size_t base = ((size_t)blockIdx.x * 8 + wave) * EMBED;

  v8f sc = {};
#pragma unroll
  for (int k0 = 0; k0 < HDIM; k0 += 32) {
    const half_t* qa = Q + base + (size_t)l16 * HDIM + k0 + hs * 16;
    v16h af = cat16(*(const v8h*)qa, *(const v8h*)(qa + 8));
    const half_t* kb = K + base + (size_t)l16 * HDIM + k0 + hs * 16;
    v16h bf = cat16(*(const v8h*)kb, *(const v8h*)(kb + 8));
    sc = __builtin_amdgcn_wmma_f32_16x16x32_f16(false, af, false, bf,
                                                (short)0, sc, false, false);
  }

  // softmax over j (j = l16, spread across lanes within each 16-lane group)
  const float scale = 0.125f;  // 1/sqrt(64)
  float w[8];
#pragma unroll
  for (int r = 0; r < 8; ++r) {
    float s = sc[r] * scale;
    float m = s;
#pragma unroll
    for (int msk = 1; msk < 16; msk <<= 1) m = fmaxf(m, __shfl_xor(m, msk, 32));
    float e = __expf(s - m);
    float sum = e;
#pragma unroll
    for (int msk = 1; msk < 16; msk <<= 1) sum += __shfl_xor(sum, msk, 32);
    w[r] = e / sum;
  }

#pragma unroll
  for (int r = 0; r < 8; ++r) wt[wave][r + hs * 8][l16] = w[r];
  __syncthreads();

  const int d0 = lane * 2;
  float vv0[16], vv1[16];
#pragma unroll
  for (int j = 0; j < 16; ++j) {
    vv0[j] = (float)V[base + (size_t)j * HDIM + d0];
    vv1[j] = (float)V[base + (size_t)j * HDIM + d0 + 1];
  }
#pragma unroll
  for (int i = 0; i < 16; ++i) {
    float a0 = 0.f, a1 = 0.f;
#pragma unroll
    for (int j = 0; j < 16; ++j) {
      const float wj = wt[wave][i][j];
      a0 += wj * vv0[j];
      a1 += wj * vv1[j];
    }
    O[base + (size_t)i * HDIM + d0]     = (half_t)a0;
    O[base + (size_t)i * HDIM + d0 + 1] = (half_t)a1;
  }
}

extern "C" void kernel_launch(void* const* d_in, const int* in_sizes, int n_in,
                              void* d_out, int out_size, void* d_ws, size_t ws_size,
                              hipStream_t stream) {
  const float* x  = (const float*)d_in[0];
  const float* wq = (const float*)d_in[1];
  const float* bq = (const float*)d_in[2];
  const float* wk = (const float*)d_in[3];
  const float* bk = (const float*)d_in[4];
  const float* wv = (const float*)d_in[5];
  const float* bv = (const float*)d_in[6];
  const float* wo = (const float*)d_in[7];
  const float* bo = (const float*)d_in[8];
  float* out = (float*)d_out;

  char* ws = (char*)d_ws;
  const size_t MB = (size_t)1 << 20;
  half_t* wqh = (half_t*)(ws + 0 * MB);     // 2 MB each
  half_t* wkh = (half_t*)(ws + 2 * MB);
  half_t* wvh = (half_t*)(ws + 4 * MB);
  half_t* woh = (half_t*)(ws + 6 * MB);
  half_t* xh  = (half_t*)(ws + 8 * MB);     // 32 MB
  half_t* qh  = (half_t*)(ws + 40 * MB);    // 32 MB
  half_t* kh  = (half_t*)(ws + 72 * MB);    // 32 MB
  half_t* vh  = (half_t*)(ws + 104 * MB);   // 32 MB
  half_t* ah  = xh;                         // xh dead after QKV; reuse for attn out

  const int NW = EMBED * EMBED;   // 1M elements
  const int NX = ROWS * EMBED;    // 16M elements
  cvt_f32_f16<<<NW / (256 * 4), 256, 0, stream>>>(wq, wqh, NW);
  cvt_f32_f16<<<NW / (256 * 4), 256, 0, stream>>>(wk, wkh, NW);
  cvt_f32_f16<<<NW / (256 * 4), 256, 0, stream>>>(wv, wvh, NW);
  cvt_f32_f16<<<NW / (256 * 4), 256, 0, stream>>>(wo, woh, NW);
  cvt_f32_f16<<<NX / (256 * 4), 256, 0, stream>>>(x, xh, NX);

  dim3 g(EMBED / 128, ROWS / 128);  // (8, 128)
  gemm16k<false><<<g, 256, 0, stream>>>(xh, wqh, bq, nullptr, qh);
  gemm16k<false><<<g, 256, 0, stream>>>(xh, wkh, bk, nullptr, kh);
  gemm16k<false><<<g, 256, 0, stream>>>(xh, wvh, bv, nullptr, vh);

  attn_heads<<<ROWS / 8, 256, 0, stream>>>(qh, kh, vh, ah);

  gemm16k<true><<<g, 256, 0, stream>>>(ah, woh, bo, out, nullptr);
}
